// SKLinear_31361851196031
// MI455X (gfx1250) — matrix operand order repaired and checked by reading, not compiled, not run on hardware
//
#include <hip/hip_runtime.h>
#include <hip/hip_bf16.h>

// SKLinear as two bf16-WMMA GEMMs:
//   T (8192x1024, bf16, ws) = X (8192x4096 f32) @ [S1|U2] (4096x1024 f32)
//   Y (8192x4096, f32)      = T @ [U1;S2] (1024x4096 f32) * 1/(2L) + bias
// B=8192, IN=4096, OUT=4096, L=4, R=128.
// Block tile 256x128, BK=32; 8 waves, each computing a 64x64 sub-tile
// (16 x v_wmma_f32_16x16x32_bf16 per K-step, 4 A-frags + 4 B-frags).

typedef __attribute__((ext_vector_type(16))) __bf16 v16bf;
typedef __attribute__((ext_vector_type(8)))  __bf16 v8bf;
typedef __attribute__((ext_vector_type(4)))  __bf16 v4bf;
typedef __attribute__((ext_vector_type(8)))  float  v8f;

#define BATCH 8192
#define IN_D  4096
#define OUT_D 4096
#define RNK   128
#define NL    4
#define KTOT  (NL * 2 * RNK)   // 1024

#define BM 256
#define BN 128
#define BK 32
#define LDSA 40   // padded row stride (elements) for As[BM][LDSA]
#define LDSB 40   // padded row stride (elements) for Bs[BN][LDSB]

// A fragment, 16x32 bf16 (ISA 7.12.2): lanes 0-15 hold row M=lane, K 0..7 (V0-3)
// and K 16..23 (V4-7); lanes 16-31 hold K 8..15 and K 24..31.
__device__ __forceinline__ v16bf load_a_frag(const __bf16* As, int row, int kh) {
  const __bf16* p = As + row * LDSA + kh * 8;
  v8bf lo = *(const v8bf*)(p);
  v8bf hi = *(const v8bf*)(p + 16);
  v16bf r;
#pragma unroll
  for (int i = 0; i < 8; ++i) { r[i] = lo[i]; r[i + 8] = hi[i]; }
  return r;
}

// B fragment, 32x16 bf16: lanes 0-15 hold col N=lane, K 0..7 and K 16..23;
// lanes 16-31 hold K 8..15 and K 24..31 (matches A).  Bs is stored [N][K].
__device__ __forceinline__ v16bf load_b_frag(const __bf16* Bs, int col, int kh) {
  const __bf16* p = Bs + col * LDSB + kh * 8;
  v8bf lo = *(const v8bf*)(p);
  v8bf hi = *(const v8bf*)(p + 16);
  v16bf r;
#pragma unroll
  for (int i = 0; i < 8; ++i) { r[i] = lo[i]; r[i + 8] = hi[i]; }
  return r;
}

// ---------------- GEMM1: T = X @ [S1|U2]  (f32 in, bf16 out) ----------------
__global__ __launch_bounds__(256) void sk_gemm1(const float* __restrict__ X,
                                                const float* __restrict__ S1,
                                                const float* __restrict__ U2,
                                                __bf16* __restrict__ T) {
  __shared__ __bf16 As[BM * LDSA];
  __shared__ __bf16 Bs[BN * LDSB];

  const int tid = threadIdx.x;
  const int m0  = blockIdx.x * BM;
  const int by  = blockIdx.y;                     // 0..7 : [S1 l=0..3 | U2 l=0..3]
  const int l   = by & 3;
  const float* Wbase = (by < 4 ? S1 : U2) + (size_t)l * IN_D * RNK;

  const int lane = tid & 31;
  const int wid  = tid >> 5;
  const int wm   = wid & 3;        // M wave group: rows wm*64 .. wm*64+63
  const int wn   = wid >> 2;       // N wave group: cols wn*64 .. wn*64+63
  const int lr   = lane & 15;
  const int kh   = lane >> 4;

  v8f acc[4][4] = {};

  for (int k0 = 0; k0 < IN_D; k0 += BK) {
    // ---- stage A tile (BM x BK f32 -> bf16): 2048 float4 slots ----
#pragma unroll
    for (int i = 0; i < 8; ++i) {
      int slot = i * 256 + tid;
      int m    = slot >> 3;                    // 8 float4 per 32-wide row
      int kq   = (slot & 7) << 2;
      const float4 v = *(const float4*)(X + (size_t)(m0 + m) * IN_D + k0 + kq);
      v4bf b;
      b[0] = (__bf16)v.x; b[1] = (__bf16)v.y; b[2] = (__bf16)v.z; b[3] = (__bf16)v.w;
      *(v4bf*)(&As[m * LDSA + kq]) = b;
      if (i == 0 && k0 + BK < IN_D)
        __builtin_prefetch((const void*)(X + (size_t)(m0 + m) * IN_D + k0 + BK + kq), 0, 1);
    }
    // ---- stage B tile (BK x BN f32 -> bf16, transposed into [N][K]) ----
#pragma unroll
    for (int i = 0; i < 4; ++i) {
      int slot = i * 256 + tid;                // 1024 slots of 4 floats
      int k    = slot >> 5;                    // 32 float4 per 128-wide row
      int nq   = (slot & 31) << 2;
      const float4 v = *(const float4*)(Wbase + (size_t)(k0 + k) * RNK + nq);
      Bs[(nq + 0) * LDSB + k] = (__bf16)v.x;
      Bs[(nq + 1) * LDSB + k] = (__bf16)v.y;
      Bs[(nq + 2) * LDSB + k] = (__bf16)v.z;
      Bs[(nq + 3) * LDSB + k] = (__bf16)v.w;
      if (i == 0 && k0 + BK < IN_D)
        __builtin_prefetch((const void*)(Wbase + (size_t)(k0 + BK + k) * RNK + nq), 0, 1);
    }
    __syncthreads();

    v16bf a[4], b[4];
#pragma unroll
    for (int mi = 0; mi < 4; ++mi) a[mi] = load_a_frag(As, wm * 64 + mi * 16 + lr, kh);
#pragma unroll
    for (int ni = 0; ni < 4; ++ni) b[ni] = load_b_frag(Bs, wn * 64 + ni * 16 + lr, kh);
#pragma unroll
    for (int mi = 0; mi < 4; ++mi)
#pragma unroll
      for (int ni = 0; ni < 4; ++ni)
        acc[mi][ni] = __builtin_amdgcn_wmma_f32_16x16x32_bf16(
            false, a[mi], false, b[ni], (short)0, acc[mi][ni], false, false);
    __syncthreads();
  }

  // ---- epilogue: bf16 T, C/D layout: N=lane&15, M = kh*8 + vgpr ----
  const int c0 = by * BN;
#pragma unroll
  for (int mi = 0; mi < 4; ++mi)
#pragma unroll
    for (int ni = 0; ni < 4; ++ni) {
      int col = c0 + wn * 64 + ni * 16 + lr;
#pragma unroll
      for (int d = 0; d < 8; ++d) {
        int row = m0 + wm * 64 + mi * 16 + kh * 8 + d;
        T[(size_t)row * KTOT + col] = (__bf16)acc[mi][ni][d];
      }
    }
}

// ---------------- GEMM2: Y = T @ [U1;S2] * 0.125 + bias ----------------
__global__ __launch_bounds__(256) void sk_gemm2(const __bf16* __restrict__ T,
                                                const float* __restrict__ U1,
                                                const float* __restrict__ S2,
                                                const float* __restrict__ bias,
                                                float* __restrict__ Y) {
  __shared__ __bf16 As[BM * LDSA];
  __shared__ __bf16 Bs[BN * LDSB];

  const int tid = threadIdx.x;
  const int m0  = blockIdx.x * BM;
  const int n0  = blockIdx.y * BN;

  const int lane = tid & 31;
  const int wid  = tid >> 5;
  const int wm   = wid & 3;
  const int wn   = wid >> 2;
  const int lr   = lane & 15;
  const int kh   = lane >> 4;

  v8f acc[4][4] = {};

  for (int k0 = 0; k0 < KTOT; k0 += BK) {
    // W2 row k0.. : first 512 rows are U1s_flat (512x4096), rest S2s_flat
    const float* Wrow = (k0 < NL * RNK) ? (U1 + (size_t)k0 * OUT_D + n0)
                                        : (S2 + (size_t)(k0 - NL * RNK) * OUT_D + n0);
    // ---- stage A tile (BM x BK bf16, direct copy): 2048 v4bf slots ----
#pragma unroll
    for (int i = 0; i < 8; ++i) {
      int slot = i * 256 + tid;
      int m    = slot >> 3;
      int kq   = (slot & 7) << 2;
      *(v4bf*)(&As[m * LDSA + kq]) =
          *(const v4bf*)(T + (size_t)(m0 + m) * KTOT + k0 + kq);
    }
    // ---- stage B tile (BK x BN f32 -> bf16, transposed) ----
#pragma unroll
    for (int i = 0; i < 4; ++i) {
      int slot = i * 256 + tid;
      int k    = slot >> 5;
      int nq   = (slot & 31) << 2;
      const float4 v = *(const float4*)(Wrow + (size_t)k * OUT_D + nq);
      Bs[(nq + 0) * LDSB + k] = (__bf16)v.x;
      Bs[(nq + 1) * LDSB + k] = (__bf16)v.y;
      Bs[(nq + 2) * LDSB + k] = (__bf16)v.z;
      Bs[(nq + 3) * LDSB + k] = (__bf16)v.w;
      if (i == 0 && k0 + BK < KTOT)
        __builtin_prefetch((const void*)(Wrow + (size_t)(BK + k) * OUT_D + nq), 0, 1);
    }
    __syncthreads();

    v16bf a[4], b[4];
#pragma unroll
    for (int mi = 0; mi < 4; ++mi) a[mi] = load_a_frag(As, wm * 64 + mi * 16 + lr, kh);
#pragma unroll
    for (int ni = 0; ni < 4; ++ni) b[ni] = load_b_frag(Bs, wn * 64 + ni * 16 + lr, kh);
#pragma unroll
    for (int mi = 0; mi < 4; ++mi)
#pragma unroll
      for (int ni = 0; ni < 4; ++ni)
        acc[mi][ni] = __builtin_amdgcn_wmma_f32_16x16x32_bf16(
            false, a[mi], false, b[ni], (short)0, acc[mi][ni], false, false);
    __syncthreads();
  }

  // ---- epilogue: scale + bias, f32 out ----
  const float scale = 1.0f / (2.0f * NL);
#pragma unroll
  for (int ni = 0; ni < 4; ++ni) {
    int col = n0 + wn * 64 + ni * 16 + lr;
    float bv = bias[col];
#pragma unroll
    for (int mi = 0; mi < 4; ++mi) {
#pragma unroll
      for (int d = 0; d < 8; ++d) {
        int row = m0 + wm * 64 + mi * 16 + kh * 8 + d;
        Y[(size_t)row * OUT_D + col] = acc[mi][ni][d] * scale + bv;
      }
    }
  }
}

extern "C" void kernel_launch(void* const* d_in, const int* in_sizes, int n_in,
                              void* d_out, int out_size, void* d_ws, size_t ws_size,
                              hipStream_t stream) {
  const float* h_in = (const float*)d_in[0];
  const float* S1s  = (const float*)d_in[1];
  const float* S2s  = (const float*)d_in[2];
  const float* U1s  = (const float*)d_in[3];
  const float* U2s  = (const float*)d_in[4];
  const float* bias = (const float*)d_in[5];
  float* out = (float*)d_out;
  __bf16* T = (__bf16*)d_ws;   // 8192 x 1024 bf16 = 16 MiB

  dim3 blk(256);
  sk_gemm1<<<dim3(BATCH / BM, KTOT / BN), blk, 0, stream>>>(h_in, S1s, U2s, T);
  sk_gemm2<<<dim3(BATCH / BM, OUT_D / BN), blk, 0, stream>>>(T, U1s, S2s, bias, out);
}
